// GRCNN_50525995270253
// MI455X (gfx1250) — matrix operand order repaired and checked by reading, not compiled, not run on hardware
//
#include <hip/hip_runtime.h>

typedef __attribute__((ext_vector_type(16))) __bf16 v16bf;
typedef __attribute__((ext_vector_type(2)))  __bf16 v2bf;
typedef __attribute__((ext_vector_type(8)))  float  v8f;

#define E_DIM   1024
#define N_NODES 1024
#define R_SRC   8192
#define H_HEADS 4
#define DH_DIM  256
#define D_MAXN  48

#define BM 128
#define BN 128
#define BK 32
#define LDS_STRIDE 40   // ushorts per row: 32 data + 8 pad (80B, 16B-aligned rows)

#if __has_builtin(__builtin_amdgcn_cvt_pk_bf16_f32)
__device__ __forceinline__ unsigned pack2(float a, float b) {
  union { v2bf v; unsigned u; } c;
  c.v = __builtin_amdgcn_cvt_pk_bf16_f32(a, b);   // v_cvt_pk_bf16_f32
  return c.u;
}
#else
// fptrunc f32->bf16 is RNE; on gfx1250 ISel should select native bf16 cvt.
__device__ __forceinline__ unsigned pack2(float a, float b) {
  union { v2bf v; unsigned u; } c;
  v2bf t = {(__bf16)a, (__bf16)b};
  c.v = t;
  return c.u;
}
#endif

union ABFrag { uint4 u2[2]; v16bf v; };

__device__ __forceinline__ void cvt_store(unsigned short* dst,
                                          float4 f0, float4 f1, float4 f2, float4 f3) {
  uint4 p0, p1;
  p0.x = pack2(f0.x, f0.y); p0.y = pack2(f0.z, f0.w);
  p0.z = pack2(f1.x, f1.y); p0.w = pack2(f1.z, f1.w);
  p1.x = pack2(f2.x, f2.y); p1.y = pack2(f2.z, f2.w);
  p1.z = pack2(f3.x, f3.y); p1.w = pack2(f3.z, f3.w);
  uint4* d4 = reinterpret_cast<uint4*>(dst);
  d4[0] = p0; d4[1] = p1;
}

// C[M,N] = act(A[M,K] @ W[N,K]^T + bias[N]); optional ReLU, optional deg==0 row zeroing.
template <bool RELU, bool MASK>
__global__ __launch_bounds__(256) void gemm_wmma(
    const float* __restrict__ A, const float* __restrict__ W,
    const float* __restrict__ bias, float* __restrict__ C,
    int M, int N, int K, const int* __restrict__ deg)
{
  __shared__ __align__(16) unsigned short As[2][BM * LDS_STRIDE];
  __shared__ __align__(16) unsigned short Bs[2][BN * LDS_STRIDE];

  const int tid  = threadIdx.x;
  const int lane = tid & 31;
  const int wave = tid >> 5;
  const int r    = lane & 15;     // tile row (A frag) / tile col (B,C,D)
  const int hs   = lane >> 4;     // lane-half select

  const int blockM = blockIdx.y * BM;
  const int blockN = blockIdx.x * BN;

  const int ldRow  = tid >> 1;    // 0..127 (tile row staged by this thread)
  const int ldHalf = tid & 1;     // which 16-float half of the 32-wide K slab

  const int wm = (wave & 1) * 64; // wave's 64-row strip
  const int wn = (wave >> 1) * 32;// wave's 32-col strip

  const float* aPtr = A + (size_t)(blockM + ldRow) * K + ldHalf * 16;
  const float* bPtr = W + (size_t)(blockN + ldRow) * K + ldHalf * 16;
  const int ldsOff  = ldRow * LDS_STRIDE + ldHalf * 16;

  const v8f vzero = {0.f, 0.f, 0.f, 0.f, 0.f, 0.f, 0.f, 0.f};
  v8f acc[4][2];
#pragma unroll
  for (int i = 0; i < 4; ++i)
#pragma unroll
    for (int j = 0; j < 2; ++j) acc[i][j] = vzero;

  // fragment compute on one LDS buffer
  auto compute = [&](const unsigned short* Ab, const unsigned short* Bb) {
    ABFrag afr[4], bfr[2];
#pragma unroll
    for (int mt = 0; mt < 4; ++mt) {
      // A 16x32 bf16 layout: lane<16 -> K{0..7,16..23}; lane>=16 -> K{8..15,24..31}
      const unsigned short* p = &Ab[(wm + mt * 16 + r) * LDS_STRIDE + hs * 8];
      afr[mt].u2[0] = *reinterpret_cast<const uint4*>(p);
      afr[mt].u2[1] = *reinterpret_cast<const uint4*>(p + 16);
    }
#pragma unroll
    for (int nt = 0; nt < 2; ++nt) {
      // B 32x16 bf16 layout: lane<16 -> K{0..7,16..23}; lane>=16 -> K{8..15,24..31}, N = lane%16
      const unsigned short* p = &Bb[(wn + nt * 16 + r) * LDS_STRIDE + hs * 8];
      bfr[nt].u2[0] = *reinterpret_cast<const uint4*>(p);
      bfr[nt].u2[1] = *reinterpret_cast<const uint4*>(p + 16);
    }
#pragma unroll
    for (int mt = 0; mt < 4; ++mt)
#pragma unroll
      for (int nt = 0; nt < 2; ++nt)
        acc[mt][nt] = __builtin_amdgcn_wmma_f32_16x16x32_bf16(
            false, afr[mt].v, false, bfr[nt].v, (short)0, acc[mt][nt], false, false);
  };

  // ---- prologue: stage slab k0=0 into buffer 0 ----
  {
    const float4* ap = reinterpret_cast<const float4*>(aPtr);
    const float4* bp = reinterpret_cast<const float4*>(bPtr);
    float4 a0 = ap[0], a1 = ap[1], a2 = ap[2], a3 = ap[3];
    float4 b0 = bp[0], b1 = bp[1], b2 = bp[2], b3 = bp[3];
    cvt_store(&As[0][ldsOff], a0, a1, a2, a3);
    cvt_store(&Bs[0][ldsOff], b0, b1, b2, b3);
  }
  __syncthreads();

  int cur = 0;
  for (int k0 = BK; k0 < K; k0 += BK) {
    // issue next-slab global loads first (latency hidden under WMMA work)
    const float4* ap = reinterpret_cast<const float4*>(aPtr + k0);
    const float4* bp = reinterpret_cast<const float4*>(bPtr + k0);
    float4 a0 = ap[0], a1 = ap[1], a2 = ap[2], a3 = ap[3];
    float4 b0 = bp[0], b1 = bp[1], b2 = bp[2], b3 = bp[3];
    if (k0 + BK < K) {
      __builtin_prefetch(aPtr + k0 + BK, 0, 0);   // global_prefetch_b8
      __builtin_prefetch(bPtr + k0 + BK, 0, 0);
    }

    compute(As[cur], Bs[cur]);

    cvt_store(&As[cur ^ 1][ldsOff], a0, a1, a2, a3);
    cvt_store(&Bs[cur ^ 1][ldsOff], b0, b1, b2, b3);
    __syncthreads();
    cur ^= 1;
  }
  compute(As[cur], Bs[cur]);

  // ---- epilogue: bias (+ReLU) (+deg mask) ----
#pragma unroll
  for (int mt = 0; mt < 4; ++mt) {
#pragma unroll
    for (int nt = 0; nt < 2; ++nt) {
      const int col = blockN + wn + nt * 16 + r;
      const float bv = bias[col];
#pragma unroll
      for (int j = 0; j < 8; ++j) {
        const int row = blockM + wm + mt * 16 + hs * 8 + j;
        float v = acc[mt][nt][j] + bv;
        if (RELU) v = fmaxf(v, 0.f);
        if (MASK && deg[row] == 0) v = 0.f;
        C[(size_t)row * N + col] = v;
      }
    }
  }
}

// Ordered compaction of adjacency row -> first D_MAXN neighbor indices + degree.
__global__ __launch_bounds__(32) void neigh_kernel(
    const float* __restrict__ adj, int* __restrict__ neigh, int* __restrict__ deg)
{
  const int n = blockIdx.x;
  const int lane = threadIdx.x;
  int base = 0;
  for (int b = 0; b < R_SRC; b += 32) {
    float a = adj[(size_t)n * R_SRC + b + lane];
    bool pred = a > 0.5f;
    unsigned m32 = (unsigned)__ballot(pred);
    if (pred) {
      int pos = base + __popc(m32 & ((1u << lane) - 1u));
      if (pos < D_MAXN) neigh[n * D_MAXN + pos] = b + lane;
    }
    base += __popc(m32);
  }
  if (lane == 0) deg[n] = base;
}

// One block per node, one wave32 per head.
__global__ __launch_bounds__(128) void attn_kernel(
    const float* __restrict__ qbuf, const float* __restrict__ Ksrc,
    const float* __restrict__ Vsrc, const int* __restrict__ neigh,
    const int* __restrict__ deg, float* __restrict__ outp)
{
  __shared__ __align__(16) float sq[H_HEADS * DH_DIM];
  __shared__ int nb[D_MAXN];

  const int n    = blockIdx.x;
  const int tid  = threadIdx.x;
  const int h    = tid >> 5;
  const int lane = tid & 31;
  const int dmax = min(deg[n], D_MAXN);

  float* orow = outp + (size_t)n * E_DIM;
  if (dmax == 0) {          // uniform per block: safe early exit, zero the row
    float4 z = make_float4(0.f, 0.f, 0.f, 0.f);
    float4* o4 = reinterpret_cast<float4*>(orow + tid * 8);
    o4[0] = z; o4[1] = z;
    return;
  }

  for (int i = tid; i < E_DIM; i += 128) sq[i] = qbuf[(size_t)n * E_DIM + i];
  if (tid < D_MAXN) nb[tid] = (tid < dmax) ? neigh[n * D_MAXN + tid] : 0;
  __syncthreads();

  const float4* q4 = reinterpret_cast<const float4*>(&sq[h * DH_DIM]);
  float s1 = -3.0e38f, s2 = -3.0e38f;
#pragma unroll
  for (int dd = 0; dd < 2; ++dd) {
    const int d = lane + dd * 32;
    if (d < dmax) {
      const float4* k4 = reinterpret_cast<const float4*>(
          Ksrc + (size_t)nb[d] * E_DIM + h * DH_DIM);
      float acc = 0.f;
#pragma unroll 8
      for (int j = 0; j < DH_DIM / 4; ++j) {
        float4 q = q4[j]; float4 k = k4[j];
        acc += q.x * k.x + q.y * k.y + q.z * k.z + q.w * k.w;
      }
      const float s = acc * 0.0625f;   // 1/sqrt(256)
      if (dd == 0) s1 = s; else s2 = s;
    }
  }
  float m = fmaxf(s1, s2);
  for (int off = 16; off > 0; off >>= 1) m = fmaxf(m, __shfl_xor(m, off, 32));
  float e1 = (lane < dmax)      ? __expf(s1 - m) : 0.f;
  float e2 = (lane + 32 < dmax) ? __expf(s2 - m) : 0.f;
  float sum = e1 + e2;
  for (int off = 16; off > 0; off >>= 1) sum += __shfl_xor(sum, off, 32);
  const float inv = 1.0f / sum;
  const float w1 = e1 * inv, w2 = e2 * inv;

  float acc[8] = {0.f, 0.f, 0.f, 0.f, 0.f, 0.f, 0.f, 0.f};
  for (int d = 0; d < dmax; ++d) {
    const float wd = (d < 32) ? __shfl(w1, d, 32) : __shfl(w2, d - 32, 32);
    const float4* v4 = reinterpret_cast<const float4*>(
        Vsrc + (size_t)nb[d] * E_DIM + h * DH_DIM + lane * 8);
    float4 va = v4[0], vb = v4[1];
    acc[0] += wd * va.x; acc[1] += wd * va.y; acc[2] += wd * va.z; acc[3] += wd * va.w;
    acc[4] += wd * vb.x; acc[5] += wd * vb.y; acc[6] += wd * vb.z; acc[7] += wd * vb.w;
  }
  float* op = orow + h * DH_DIM + lane * 8;
  reinterpret_cast<float4*>(op)[0] = make_float4(acc[0], acc[1], acc[2], acc[3]);
  reinterpret_cast<float4*>(op)[1] = make_float4(acc[4], acc[5], acc[6], acc[7]);
}

extern "C" void kernel_launch(void* const* d_in, const int* in_sizes, int n_in,
                              void* d_out, int out_size, void* d_ws, size_t ws_size,
                              hipStream_t stream) {
  const float* target  = (const float*)d_in[0];
  const float* source  = (const float*)d_in[1];
  const float* adj     = (const float*)d_in[2];
  const float* W_trans = (const float*)d_in[3];
  const float* b_trans = (const float*)d_in[4];
  const float* in_w    = (const float*)d_in[5];
  const float* in_b    = (const float*)d_in[6];
  const float* out_w   = (const float*)d_in[7];
  const float* out_b   = (const float*)d_in[8];
  float* out = (float*)d_out;

  float* src_relu = (float*)d_ws;                                 // [R,E]
  float* Ksrc = src_relu + (size_t)R_SRC * E_DIM;                 // [R,E]
  float* Vsrc = Ksrc + (size_t)R_SRC * E_DIM;                     // [R,E]
  float* qbuf = Vsrc + (size_t)R_SRC * E_DIM;                     // [N,E]
  float* attn = qbuf + (size_t)N_NODES * E_DIM;                   // [N,E]
  int* neigh  = (int*)(attn + (size_t)N_NODES * E_DIM);           // [N,48]
  int* deg    = neigh + N_NODES * D_MAXN;                         // [N]

  const float* Wq = in_w;
  const float* Wk = in_w + (size_t)E_DIM * E_DIM;
  const float* Wv = in_w + 2 * (size_t)E_DIM * E_DIM;
  const float* bq = in_b;
  const float* bk = in_b + E_DIM;
  const float* bv = in_b + 2 * E_DIM;

  const dim3 blk(256);
  const dim3 gridR(E_DIM / BN, R_SRC / BM);
  const dim3 gridN(E_DIM / BN, N_NODES / BM);

  gemm_wmma<true,  false><<<gridR, blk, 0, stream>>>(source,   W_trans, b_trans, src_relu, R_SRC,   E_DIM, E_DIM, nullptr);
  gemm_wmma<false, false><<<gridR, blk, 0, stream>>>(src_relu, Wk,      bk,      Ksrc,     R_SRC,   E_DIM, E_DIM, nullptr);
  gemm_wmma<false, false><<<gridR, blk, 0, stream>>>(src_relu, Wv,      bv,      Vsrc,     R_SRC,   E_DIM, E_DIM, nullptr);
  gemm_wmma<false, false><<<gridN, blk, 0, stream>>>(target,   Wq,      bq,      qbuf,     N_NODES, E_DIM, E_DIM, nullptr);
  neigh_kernel<<<N_NODES, 32, 0, stream>>>(adj, neigh, deg);
  attn_kernel<<<N_NODES, 128, 0, stream>>>(qbuf, Ksrc, Vsrc, neigh, deg, attn);
  gemm_wmma<false, true ><<<gridN, blk, 0, stream>>>(attn,     out_w,   out_b,   out,      N_NODES, E_DIM, E_DIM, deg);
}
